// FeatureSpacePurityEntropyScore_4166118277853
// MI455X (gfx1250) — matrix operand order repaired and checked by reading, not compiled, not run on hardware
//
#include <hip/hip_runtime.h>

typedef __attribute__((ext_vector_type(8)))  _Float16 v8h;
typedef __attribute__((ext_vector_type(16))) _Float16 v16h;
typedef __attribute__((ext_vector_type(8)))  float    v8f;

#define NS      32400      // sampled pixels
#define CDIM    512        // feature channels
#define HW      129600     // 270*480
#define NCLS    19
#define FULLPIX 2073600    // 1080*1920
#define N_PAD   32512      // 254 * 128, >= COL_END
#define BM      128        // query rows per workgroup (1 thread / row, 4 waves)
#define BN      64         // candidate cols per tile (4 wmma tiles wide)
#define KNN     9
#define COL_END 32448      // 507 * 64 tiles cover all 32400 cols

// ---------------------------------------------------------------------------
// Prep: gather sampled pixels -> row-major f16 features + fp32 sq-norms + cls
// ---------------------------------------------------------------------------
__global__ __launch_bounds__(256) void prep_kernel(
    const float* __restrict__ feat,      // [512, 129600] (C,H*W)
    const float* __restrict__ outp,      // [19, 129600]
    const int*   __restrict__ sidx,      // [NS]
    _Float16*    __restrict__ F,         // [N_PAD, 512] f16 (pre-zeroed)
    float*       __restrict__ sq,        // [N_PAD] fp32 (pre-zeroed)
    int*         __restrict__ cls)       // [NS]
{
  const int row = blockIdx.x;            // 0..NS-1
  const int s   = sidx[row];
  const int tid = threadIdx.x;
  __shared__ float red[256];

  float acc = 0.0f;
  for (int c = tid; c < CDIM; c += 256) {
    float v = feat[(size_t)c * HW + s];
    F[(size_t)row * CDIM + c] = (_Float16)v;
    acc += v * v;
  }
  red[tid] = acc;
  __syncthreads();
  for (int off = 128; off > 0; off >>= 1) {
    if (tid < off) red[tid] += red[tid + off];
    __syncthreads();
  }
  if (tid == 0) {
    sq[row] = red[0];
    float best = outp[s];
    int bc = 0;
    for (int c = 1; c < NCLS; ++c) {
      float v = outp[(size_t)c * HW + s];
      if (v > best) { best = v; bc = c; }     // strict > keeps first max (argmax)
    }
    cls[row] = bc;
  }
}

// ---------------------------------------------------------------------------
// Prototype squared norms
// ---------------------------------------------------------------------------
__global__ void proto_kernel(const float* __restrict__ protos,
                             float* __restrict__ psq)
{
  const int c = threadIdx.x;
  if (c < NCLS) {
    const float* p = protos + (size_t)c * CDIM;
    float s = 0.0f;
    for (int k = 0; k < CDIM; ++k) s += p[k] * p[k];
    psq[c] = s;
  }
}

// ---------------------------------------------------------------------------
// Main: WMMA Gram blocks -> distances in LDS -> fused per-row top-9 ->
//       class entropy + prototype rank -> scatter into full-res planes.
// ---------------------------------------------------------------------------
__global__ __launch_bounds__(128) void knn_purity_kernel(
    const _Float16* __restrict__ F,      // [N_PAD, 512] f16
    const float*    __restrict__ sq,     // [N_PAD] squared norms
    const int*      __restrict__ cls,    // [NS]
    const float*    __restrict__ protos, // [19, 512]
    const float*    __restrict__ psq,    // [19]
    const int*      __restrict__ sidx,   // [NS]
    float*          __restrict__ out)    // [2, 1080*1920]
{
  __shared__ float dls[BM * 65];         // 128 x 64 distances, stride 65 (conflict-free)

  const int tid  = threadIdx.x;
  const int wv   = tid >> 5;             // 0..3
  const int lane = tid & 31;
  const int m    = lane & 15;
  const int half = lane >> 4;
  const int row0 = blockIdx.x * BM;

  // q_sq fragment for this wave's two 16-row C tiles (D layout: M = v + 8*half)
  float qsr[2][8];
#pragma unroll
  for (int t = 0; t < 2; ++t) {
    const int base = row0 + (wv * 2 + t) * 16 + half * 8;
#pragma unroll
    for (int v = 0; v < 8; ++v) qsr[t][v] = sq[base + v];
  }

  float nd[KNN];
  int   ni[KNN];
#pragma unroll
  for (int k = 0; k < KNN; ++k) { nd[k] = 3.0e38f; ni[k] = 0; }

  for (int c0 = 0; c0 < COL_END; c0 += BN) {
    v8f acc[2][4];
#pragma unroll
    for (int t = 0; t < 2; ++t)
#pragma unroll
      for (int u = 0; u < 4; ++u) acc[t][u] = (v8f)0.0f;

    for (int kk = 0; kk < CDIM; kk += 32) {
      // A fragments (16x32 f16): half-lane 0 holds K {0..7,16..23}, half 1 {8..15,24..31}
      v16h afr[2];
#pragma unroll
      for (int t = 0; t < 2; ++t) {
        const _Float16* p =
            F + (size_t)(row0 + (wv * 2 + t) * 16 + m) * CDIM + kk + half * 8;
        v8h lo = *(const v8h*)p;          // K kk+h*8 .. +7
        v8h hi = *(const v8h*)(p + 16);   // K kk+16+h*8 .. +7
        afr[t] = __builtin_shufflevector(lo, hi,
                 0,1,2,3,4,5,6,7,8,9,10,11,12,13,14,15);
      }
      // B fragments (32x16 f16): B^T is the same row-major matrix;
      // half-lane 0 holds K {0..7,16..23}, half 1 {8..15,24..31}
      v16h bfr[4];
#pragma unroll
      for (int u = 0; u < 4; ++u) {
        const _Float16* p =
            F + (size_t)(c0 + u * 16 + m) * CDIM + kk + half * 8;
        v8h lo = *(const v8h*)p;
        v8h hi = *(const v8h*)(p + 16);
        bfr[u] = __builtin_shufflevector(lo, hi,
                 0,1,2,3,4,5,6,7,8,9,10,11,12,13,14,15);
      }
#pragma unroll
      for (int t = 0; t < 2; ++t)
#pragma unroll
        for (int u = 0; u < 4; ++u)
          acc[t][u] = __builtin_amdgcn_wmma_f32_16x16x32_f16(
              false, afr[t], false, bfr[u], (short)0, acc[t][u], false, false);
    }

    // distance epilogue -> LDS: d = |q|^2 + |c|^2 - 2*dot
#pragma unroll
    for (int t = 0; t < 2; ++t) {
      const int lr0 = (wv * 2 + t) * 16 + half * 8;
#pragma unroll
      for (int u = 0; u < 4; ++u) {
        const int ln = u * 16 + m;
        const float cq = sq[c0 + ln];
#pragma unroll
        for (int v = 0; v < 8; ++v)
          dls[(lr0 + v) * 65 + ln] = qsr[t][v] + cq - 2.0f * acc[t][u][v];
      }
    }
    __syncthreads();

    // fused top-9 scan: one thread per query row
    const int grow = row0 + tid;
    if (grow < NS) {
      for (int j = 0; j < BN; ++j) {
        const int col = c0 + j;
        float d = fmaxf(dls[tid * 65 + j], 1e-12f);
        if (col < NS && d < nd[KNN - 1]) {
          nd[KNN - 1] = d; ni[KNN - 1] = col;
#pragma unroll
          for (int k = KNN - 1; k > 0; --k) {
            if (nd[k] < nd[k - 1]) {
              float td = nd[k]; nd[k] = nd[k - 1]; nd[k - 1] = td;
              int   ti = ni[k]; ni[k] = ni[k - 1]; ni[k - 1] = ti;
            }
          }
        }
      }
    }
    __syncthreads();
  }

  const int grow = row0 + tid;
  if (grow >= NS) return;

  // purity: entropy of the 9-NN class histogram
  int ncl[KNN];
#pragma unroll
  for (int k = 0; k < KNN; ++k) ncl[k] = cls[ni[k]];
  float ent = 0.0f;
  for (int c = 0; c < NCLS; ++c) {
    int cnt = 0;
#pragma unroll
    for (int k = 0; k < KNN; ++k) cnt += (ncl[k] == c) ? 1 : 0;
    if (cnt) {
      float p = (float)cnt * (1.0f / 9.0f);
      ent -= p * __logf(p + 1e-6f);
    }
  }
  ent *= 0.33962347f;                     // 1 / ln(19)

  // disparity: rank of predicted class among prototype distances (stable argsort)
  const _Float16* q = F + (size_t)grow * CDIM;
  const float myq = sq[grow];
  const int   myc = cls[grow];
  float dpm;
  {
    const float* pr = protos + (size_t)myc * CDIM;
    float s = 0.0f;
    for (int k = 0; k < CDIM; k += 8) {
      v8h qq = *(const v8h*)(q + k);
#pragma unroll
      for (int x = 0; x < 8; ++x) s += (float)qq[x] * pr[k + x];
    }
    dpm = fmaxf(myq - 2.0f * s + psq[myc], 1e-12f);
  }
  int rank = 0;
  for (int c = 0; c < NCLS; ++c) {
    if (c == myc) continue;
    const float* pr = protos + (size_t)c * CDIM;
    float s = 0.0f;
    for (int k = 0; k < CDIM; k += 8) {
      v8h qq = *(const v8h*)(q + k);
#pragma unroll
      for (int x = 0; x < 8; ++x) s += (float)qq[x] * pr[k + x];
    }
    float dc = fmaxf(myq - 2.0f * s + psq[c], 1e-12f);
    if (dc < dpm || (dc == dpm && c < myc)) ++rank;
  }
  const float score = (float)rank * (1.0f / 18.0f);

  const int s = sidx[grow];
  out[s]           = ent;                 // plane 0: purity (entropy values)
  out[FULLPIX + s] = score;               // plane 1: disparity score
}

// ---------------------------------------------------------------------------
extern "C" void kernel_launch(void* const* d_in, const int* in_sizes, int n_in,
                              void* d_out, int out_size, void* d_ws, size_t ws_size,
                              hipStream_t stream) {
  (void)in_sizes; (void)n_in; (void)out_size; (void)ws_size;

  const float* feat   = (const float*)d_in[0];   // [1,512,270,480]
  const float* outp   = (const float*)d_in[1];   // [1,19,270,480]
  const float* protos = (const float*)d_in[2];   // [1,19,512]
  const int*   sidx   = (const int*)d_in[3];     // [32400]
  float* out = (float*)d_out;                    // [2,1080,1920]

  // workspace layout (contiguous): f16 feats | sq-norms | classes | proto norms
  char* w = (char*)d_ws;
  _Float16* F   = (_Float16*)w;                                   // N_PAD*512 f16
  float*    sq  = (float*)(w + (size_t)N_PAD * CDIM * 2);         // N_PAD fp32
  int*      cls = (int*)((char*)sq + (size_t)N_PAD * 4);          // NS int
  float*    psq = (float*)((char*)cls + (size_t)NS * 4);          // 19 fp32

  // zero f16 feats + sq (padding rows must be zero) and the output planes
  hipMemsetAsync(F, 0, (size_t)N_PAD * CDIM * 2 + (size_t)N_PAD * 4, stream);
  hipMemsetAsync(out, 0, (size_t)2 * FULLPIX * 4, stream);

  prep_kernel<<<NS, 256, 0, stream>>>(feat, outp, sidx, F, sq, cls);
  proto_kernel<<<1, 32, 0, stream>>>(protos, psq);

  const int nblk = (NS + BM - 1) / BM;   // 254
  knn_purity_kernel<<<nblk, BM, 0, stream>>>(F, sq, cls, protos, psq, sidx, out);
}
